// MultiHeadAttention_60155311948194
// MI455X (gfx1250) — matrix operand order repaired and checked by reading, not compiled, not run on hardware
//
#include <hip/hip_runtime.h>

#define BATCH 4
#define SEQ 2048
#define DMODEL 256
#define NHEAD 4
#define HDIM 64
#define S2 2052   // padded LDS row stride (floats): 2052 % 64 == 4 -> conflict-free column reads

typedef __attribute__((ext_vector_type(16))) __bf16 v16bf;
typedef __attribute__((ext_vector_type(8)))  float v8f;
typedef __attribute__((ext_vector_type(8)))  unsigned short ushort8;

union Frag {
    v16bf v;
    ushort8 h[2];
    unsigned short u[16];
};

__device__ __forceinline__ unsigned short f32_bf16(float f) {
    unsigned int u = __float_as_uint(f);
    unsigned int r = u + 0x7FFFu + ((u >> 16) & 1u);   // round-to-nearest-even
    return (unsigned short)(r >> 16);
}

__device__ __forceinline__ v8f wmma_bf16(const Frag& a, const Frag& b, v8f c) {
    return __builtin_amdgcn_wmma_f32_16x16x32_bf16(false, a.v, false, b.v,
                                                   (short)0, c, false, false);
}

// ---------------- conversion kernels ----------------
__global__ void k_cvt_x(const float* __restrict__ src, unsigned short* __restrict__ dst, int n) {
    int i = blockIdx.x * blockDim.x + threadIdx.x;
    if (i < n) dst[i] = f32_bf16(src[i]);
}

// Wt[w][n][k] = bf16(W_w[k][n])  (transposed so B-fragment K-dim is contiguous)
__global__ void k_cvt_wt(const float* __restrict__ Wq, const float* __restrict__ Wk,
                         const float* __restrict__ Wv, const float* __restrict__ Wo,
                         unsigned short* __restrict__ dst) {
    int i = blockIdx.x * blockDim.x + threadIdx.x;
    int w = blockIdx.y;
    const float* W = (w == 0) ? Wq : (w == 1) ? Wk : (w == 2) ? Wv : Wo;
    int n = i >> 8, k = i & 255;
    dst[w * (DMODEL * DMODEL) + n * DMODEL + k] = f32_bf16(W[k * DMODEL + n]);
}

// ---------------- QKV projection GEMM: C[8192,256] = Xb * W  ----------------
// block = 8 waves; wave computes a 16x64 strip; z selects Q/K/V.
__global__ void __launch_bounds__(256) k_gemm_qkv(
        const unsigned short* __restrict__ Xb, const unsigned short* __restrict__ Wt,
        const float* __restrict__ bq, const float* __restrict__ bk, const float* __restrict__ bv,
        unsigned short* __restrict__ Qb, unsigned short* __restrict__ Kb,
        unsigned short* __restrict__ Vt) {
    const int lane = threadIdx.x & 31;
    const int wave = threadIdx.x >> 5;
    const int mlo = lane & 15, hi = lane >> 4;
    const int z = blockIdx.z;
    const int m0 = blockIdx.x * 128 + wave * 16;
    const int n0 = blockIdx.y * 64;
    const unsigned short* W = Wt + z * (DMODEL * DMODEL);
    const float* bias = (z == 0) ? bq : (z == 1) ? bk : bv;

    v8f acc[4] = {};
    const unsigned short* arow = Xb + (m0 + mlo) * DMODEL;
    for (int kk = 0; kk < DMODEL; kk += 32) {
        Frag af;
        af.h[0] = *(const ushort8*)(arow + kk + 8 * hi);
        af.h[1] = *(const ushort8*)(arow + kk + 16 + 8 * hi);
#pragma unroll
        for (int t = 0; t < 4; ++t) {
            const unsigned short* bcol = W + (n0 + t * 16 + mlo) * DMODEL + kk + 8 * hi;
            Frag bf;
            bf.h[0] = *(const ushort8*)(bcol);
            bf.h[1] = *(const ushort8*)(bcol + 16);
            acc[t] = wmma_bf16(af, bf, acc[t]);
        }
    }
#pragma unroll
    for (int t = 0; t < 4; ++t)
#pragma unroll
        for (int r = 0; r < 8; ++r) {
            int row = m0 + r + 8 * hi;
            int col = n0 + t * 16 + mlo;
            unsigned short v = f32_bf16(acc[t][r] + bias[col]);
            if (z == 0)      Qb[row * DMODEL + col] = v;
            else if (z == 1) Kb[row * DMODEL + col] = v;
            else {           // V stored head-transposed: [b][h][hd][s]
                int b = row >> 11, s = row & (SEQ - 1);
                int hh = col >> 6, hd = col & (HDIM - 1);
                Vt[((b * NHEAD + hh) * HDIM + hd) * SEQ + s] = v;
            }
        }
}

// ---------------- fused attention: scores -> softmax -> attn out -> P*V ----------------
// 4 waves per block share one 16-row x 2048-col score strip in LDS (128 KB).
// Work split avoids any cross-wave reduction:
//   Phase A: wave w computes key tiles [w*32, w*32+32)
//   Phase B: wave w softmaxes rows [w*4, w*4+4) (stats wave-local via shfl)
//   Phase C: wave w computes V column tile w (full K loop, no partials)
__global__ void __launch_bounds__(128) k_attn(
        const unsigned short* __restrict__ Qb, const unsigned short* __restrict__ Kb,
        const unsigned short* __restrict__ Vt, unsigned short* __restrict__ Ctxb,
        float* __restrict__ attn) {
    extern __shared__ float sP[];                 // 16 x S2 floats
    const int lane = threadIdx.x & 31;
    const int wave = threadIdx.x >> 5;            // 0..3
    const int mlo = lane & 15, hi = lane >> 4;
    const int sq0 = blockIdx.x * 16;
    const int b = blockIdx.y >> 2, h = blockIdx.y & 3;

    // Q fragments (full K = 64, two K=32 fragments), loaded once per wave
    const unsigned short* qrow = Qb + (b * SEQ + sq0 + mlo) * DMODEL + h * HDIM;
    Frag aq0, aq1;
    aq0.h[0] = *(const ushort8*)(qrow + 8 * hi);
    aq0.h[1] = *(const ushort8*)(qrow + 16 + 8 * hi);
    aq1.h[0] = *(const ushort8*)(qrow + 32 + 8 * hi);
    aq1.h[1] = *(const ushort8*)(qrow + 48 + 8 * hi);

    // Phase A: S = Q K^T / sqrt(HD), into LDS (wave-private key-tile range)
    const unsigned short* kbase = Kb + (b * SEQ) * DMODEL + h * HDIM;
    for (int nt = wave * 32; nt < wave * 32 + 32; ++nt) {
        const unsigned short* krow = kbase + (nt * 16 + mlo) * DMODEL;
        Frag bk0, bk1;
        bk0.h[0] = *(const ushort8*)(krow + 8 * hi);
        bk0.h[1] = *(const ushort8*)(krow + 16 + 8 * hi);
        bk1.h[0] = *(const ushort8*)(krow + 32 + 8 * hi);
        bk1.h[1] = *(const ushort8*)(krow + 48 + 8 * hi);
        v8f acc = {};
        acc = wmma_bf16(aq0, bk0, acc);
        acc = wmma_bf16(aq1, bk1, acc);
#pragma unroll
        for (int r = 0; r < 8; ++r)
            sP[(r + 8 * hi) * S2 + nt * 16 + mlo] = acc[r] * 0.125f;  // 1/sqrt(64)
    }
    __syncthreads();

    // Phase B: row softmax; wave w owns rows w*4..w*4+3, 32 lanes cooperate per row
    float rsum[4];
#pragma unroll
    for (int rr = 0; rr < 4; ++rr) {
        const int r = wave * 4 + rr;
        float* prow = sP + r * S2;
        float mx = -3.402823e38f;
        for (int c = lane; c < SEQ; c += 32) mx = fmaxf(mx, prow[c]);
#pragma unroll
        for (int m = 16; m >= 1; m >>= 1) mx = fmaxf(mx, __shfl_xor(mx, m));
        float s = 0.f;
        for (int c = lane; c < SEQ; c += 32) {
            float e = __expf(prow[c] - mx);
            prow[c] = e;
            s += e;
        }
#pragma unroll
        for (int m = 16; m >= 1; m >>= 1) s += __shfl_xor(s, m);
        rsum[rr] = s;
    }

    // normalize own rows in LDS + single coalesced write of attn probabilities
    float* attnBase = attn + (((size_t)(b * NHEAD + h) * SEQ) + sq0) * SEQ;
#pragma unroll
    for (int rr = 0; rr < 4; ++rr) {
        const int r = wave * 4 + rr;
        float rs = 1.0f / rsum[rr];
        float4* src = (float4*)(sP + r * S2);
        float4* dst = (float4*)(attnBase + (size_t)r * SEQ);
        for (int c4 = lane; c4 < SEQ / 4; c4 += 32) {
            float4 v = src[c4];
            v.x *= rs; v.y *= rs; v.z *= rs; v.w *= rs;
            src[c4] = v;
            dst[c4] = v;
        }
    }
    __syncthreads();

    // Phase C: ctx = P * V  (wave w owns V column tile w; P from LDS f32 -> bf16)
    v8f ctx = {};
    const unsigned short* vbase = Vt + (size_t)(b * NHEAD + h) * HDIM * SEQ;
    for (int kt = 0; kt < SEQ / 32; ++kt) {
        Frag af;
#pragma unroll
        for (int e = 0; e < 16; ++e) {
            int k = kt * 32 + ((e & 8) ? 16 : 0) + 8 * hi + (e & 7);
            af.u[e] = f32_bf16(sP[mlo * S2 + k]);
        }
        const unsigned short* vrow = vbase + (wave * 16 + mlo) * SEQ + kt * 32 + 8 * hi;
        Frag bf;
        bf.h[0] = *(const ushort8*)(vrow);
        bf.h[1] = *(const ushort8*)(vrow + 16);
        ctx = wmma_bf16(af, bf, ctx);
    }
#pragma unroll
    for (int r = 0; r < 8; ++r) {
        int row = b * SEQ + sq0 + r + 8 * hi;
        int col = h * HDIM + wave * 16 + mlo;
        Ctxb[row * DMODEL + col] = f32_bf16(ctx[r]);
    }
}

// ---------------- output projection + bias + residual -> Y (f32) ----------------
__global__ void __launch_bounds__(256) k_gemm_out(
        const unsigned short* __restrict__ Ctxb, const unsigned short* __restrict__ Wot,
        const float* __restrict__ bo, const float* __restrict__ x, float* __restrict__ Y) {
    const int lane = threadIdx.x & 31;
    const int wave = threadIdx.x >> 5;
    const int mlo = lane & 15, hi = lane >> 4;
    const int m0 = blockIdx.x * 128 + wave * 16;
    const int n0 = blockIdx.y * 64;

    v8f acc[4] = {};
    const unsigned short* arow = Ctxb + (m0 + mlo) * DMODEL;
    for (int kk = 0; kk < DMODEL; kk += 32) {
        Frag af;
        af.h[0] = *(const ushort8*)(arow + kk + 8 * hi);
        af.h[1] = *(const ushort8*)(arow + kk + 16 + 8 * hi);
#pragma unroll
        for (int t = 0; t < 4; ++t) {
            const unsigned short* bcol = Wot + (n0 + t * 16 + mlo) * DMODEL + kk + 8 * hi;
            Frag bf;
            bf.h[0] = *(const ushort8*)(bcol);
            bf.h[1] = *(const ushort8*)(bcol + 16);
            acc[t] = wmma_bf16(af, bf, acc[t]);
        }
    }
#pragma unroll
    for (int t = 0; t < 4; ++t)
#pragma unroll
        for (int r = 0; r < 8; ++r) {
            int row = m0 + r + 8 * hi;
            int col = n0 + t * 16 + mlo;
            Y[row * DMODEL + col] = acc[t][r] + bo[col] + x[row * DMODEL + col];
        }
}

// ---------------- LayerNorm: one wave per row of 256 ----------------
__global__ void __launch_bounds__(256) k_ln(const float* __restrict__ Y,
                                            const float* __restrict__ gamma,
                                            const float* __restrict__ beta,
                                            float* __restrict__ out) {
    const int lane = threadIdx.x & 31;
    const int wave = threadIdx.x >> 5;
    const int row = blockIdx.x * 8 + wave;
    const float* y = Y + row * DMODEL;
    float v[8], s = 0.f, s2 = 0.f;
#pragma unroll
    for (int i = 0; i < 8; ++i) {
        v[i] = y[lane + 32 * i];
        s += v[i];
        s2 += v[i] * v[i];
    }
#pragma unroll
    for (int m = 16; m >= 1; m >>= 1) {
        s += __shfl_xor(s, m);
        s2 += __shfl_xor(s2, m);
    }
    float mean = s * (1.0f / 256.0f);
    float var = s2 * (1.0f / 256.0f) - mean * mean;
    float rinv = rsqrtf(var + 1e-5f);
#pragma unroll
    for (int i = 0; i < 8; ++i) {
        int c = lane + 32 * i;
        out[row * DMODEL + c] = (v[i] - mean) * rinv * gamma[c] + beta[c];
    }
}

extern "C" void kernel_launch(void* const* d_in, const int* in_sizes, int n_in,
                              void* d_out, int out_size, void* d_ws, size_t ws_size,
                              hipStream_t stream) {
    (void)in_sizes; (void)n_in; (void)out_size; (void)ws_size;
    const float* x  = (const float*)d_in[0];
    const float* Wq = (const float*)d_in[1];
    const float* bq = (const float*)d_in[2];
    const float* Wk = (const float*)d_in[3];
    const float* bk = (const float*)d_in[4];
    const float* Wv = (const float*)d_in[5];
    const float* bv = (const float*)d_in[6];
    const float* Wo = (const float*)d_in[7];
    const float* bo = (const float*)d_in[8];
    const float* gamma = (const float*)d_in[9];
    const float* beta  = (const float*)d_in[10];

    float* out  = (float*)d_out;
    float* attn = out + (size_t)BATCH * SEQ * DMODEL;   // outputs concatenated: out, attn

    const size_t NXE = (size_t)BATCH * SEQ * DMODEL;    // 2,097,152 elements
    unsigned short* Xb   = (unsigned short*)d_ws;       // bf16 x
    unsigned short* Wt   = Xb + NXE;                    // 4 transposed bf16 weight mats
    unsigned short* Qb   = Wt + 4 * DMODEL * DMODEL;    // bf16 Q [8192][256]
    unsigned short* Kb   = Qb + NXE;                    // bf16 K [8192][256]
    unsigned short* Vt   = Kb + NXE;                    // bf16 V [b][h][hd][s]
    unsigned short* Ctxb = Vt + NXE;                    // bf16 ctx [8192][256]
    float*          Yf   = (float*)(Ctxb + NXE);        // f32 pre-LN

    k_cvt_x<<<dim3(NXE / 256), 256, 0, stream>>>(x, Xb, (int)NXE);
    k_cvt_wt<<<dim3((DMODEL * DMODEL) / 256, 4), 256, 0, stream>>>(Wq, Wk, Wv, Wo, Wt);
    k_gemm_qkv<<<dim3(64, 4, 3), 256, 0, stream>>>(Xb, Wt, bq, bk, bv, Qb, Kb, Vt);
    k_attn<<<dim3(SEQ / 16, BATCH * NHEAD), 128, 16 * S2 * sizeof(float), stream>>>(
        Qb, Kb, Vt, Ctxb, attn);
    k_gemm_out<<<dim3(64, 4), 256, 0, stream>>>(Ctxb, Wt + 3 * DMODEL * DMODEL, bo, x, Yf);
    k_ln<<<dim3(1024), 256, 0, stream>>>(Yf, gamma, beta, out);
}
